// MoE_22986664968197
// MI455X (gfx1250) — compile-verified, matrix operand order checked by reading, not run on hardware
//
#include <hip/hip_runtime.h>
#include <hip/hip_bf16.h>
#include <math.h>

// ---------------------------------------------------------------------------
// Types for CDNA5 WMMA bf16 path
// ---------------------------------------------------------------------------
typedef __bf16 bf16_t;
typedef __attribute__((ext_vector_type(16))) __bf16 v16bf;
typedef __attribute__((ext_vector_type(8)))  __bf16 v8bf;
typedef __attribute__((ext_vector_type(8)))  float  v8f;

union BF16x16 { v16bf v; v8bf h[2]; };

// Problem constants (match reference)
#define NTOK   2048          // B*S
#define HID_   2048
#define MOEI   768
#define NEXP   32
#define TOPK   8
#define NASSIGN (NTOK * TOPK)   // 16384 compacted rows (exact)

// Workspace layout (bytes)
#define WS_CW       0                               // 2048*32 f32   = 262144
#define WS_COUNTS   262144                          // 32 int
#define WS_OFFSETS  (262144 + 128)                  // 33 int
#define WS_TOKIDS   263168                          // 16384 int     = 65536
#define WS_WTS      328704                          // 16384 f32     = 65536
#define WS_H        394240                          // 16384*768 bf16 = 25165824
#define WS_HS       (394240 + 25165824)             // 2048*768 bf16  = 3145728

// ---------------------------------------------------------------------------
// Fragment helpers.
// 16-bit A (16x32) / B (32x16) layout: lane l (l<16) holds row/col l with
// K = {0..7} in regs 0..3 and K = {16..23} in regs 4..7; lanes >=16 hold the
// K = {8..15}, {24..31} halves.  So a fragment = two contiguous 16B chunks at
// base + khalf and base + khalf + 16 (elements).
// ---------------------------------------------------------------------------
__device__ __forceinline__ v16bf load_frag_f32(const float* p) {
  v8f lo = *(const v8f*)(p);
  v8f hi = *(const v8f*)(p + 16);
  BF16x16 r;
  r.h[0] = __builtin_convertvector(lo, v8bf);
  r.h[1] = __builtin_convertvector(hi, v8bf);
  return r.v;
}

__device__ __forceinline__ v16bf load_frag_bf16(const bf16_t* p) {
  BF16x16 r;
  r.h[0] = *(const v8bf*)(p);
  r.h[1] = *(const v8bf*)(p + 16);
  return r.v;
}

__device__ __forceinline__ v8f wmma_bf16(v16bf a, v16bf b, v8f c) {
  return __builtin_amdgcn_wmma_f32_16x16x32_bf16(false, a, false, b,
                                                 (short)0, c, false, false);
}

__device__ __forceinline__ float silu_f(float g) {
  return g / (1.0f + __expf(-g));
}

// ---------------------------------------------------------------------------
// Kernel 1: router.  One wave per token; lane e computes logit for expert e,
// lane 0 performs sigmoid + group-top2 -> top-4 groups -> top-8 experts.
// Emits combine weights cw[token][expert] (0 for unselected experts).
// ---------------------------------------------------------------------------
__global__ void __launch_bounds__(32)
k_router(const float* __restrict__ x, const float* __restrict__ rw,
         const float* __restrict__ bias, float* __restrict__ cw) {
  const int t = blockIdx.x;
  const int lane = threadIdx.x;             // 0..31 == expert id
  const float* xr = x + (size_t)t * HID_;
  const float* wr = rw + (size_t)lane * HID_;
  float acc = 0.f;
  for (int h = 0; h < HID_; h += 4) {
    float4 xv = *(const float4*)(xr + h);
    float4 wv = *(const float4*)(wr + h);
    acc += xv.x * wv.x + xv.y * wv.y + xv.z * wv.z + xv.w * wv.w;
  }
  __shared__ float sc[NEXP];
  sc[lane] = acc;
  __syncthreads();
  if (lane == 0) {
    float scores[NEXP], sfc[NEXP];
    for (int e = 0; e < NEXP; ++e) {
      float s = 1.0f / (1.0f + __expf(-sc[e]));
      scores[e] = s;
      sfc[e] = s + bias[e];
    }
    // group score = sum of top-2 within each group of 4
    float gs[8];
    for (int g = 0; g < 8; ++g) {
      float m1 = -1e30f, m2 = -1e30f;
      for (int i = 0; i < 4; ++i) {
        float v = sfc[g * 4 + i];
        if (v > m1)      { m2 = m1; m1 = v; }
        else if (v > m2) { m2 = v; }
      }
      gs[g] = m1 + m2;
    }
    bool gsel[8] = {false, false, false, false, false, false, false, false};
    for (int it = 0; it < 4; ++it) {
      int bj = 0; float bv = -1e30f;
      for (int g = 0; g < 8; ++g)
        if (!gsel[g] && gs[g] > bv) { bv = gs[g]; bj = g; }
      gsel[bj] = true;
    }
    float ms[NEXP];
    for (int e = 0; e < NEXP; ++e) ms[e] = gsel[e >> 2] ? sfc[e] : 0.0f;
    int idx[TOPK]; float w[TOPK]; float wsum = 0.f;
    for (int it = 0; it < TOPK; ++it) {
      int bj = 0; float bv = -1e30f;
      for (int e = 0; e < NEXP; ++e)
        if (ms[e] > bv) { bv = ms[e]; bj = e; }
      ms[bj] = -1e30f;
      idx[it] = bj;
      w[it] = scores[bj];          // weight from sigmoid score (no bias)
      wsum += w[it];
    }
    float row[NEXP];
    for (int e = 0; e < NEXP; ++e) row[e] = 0.f;
    const float scale = 2.5f / (wsum + 1e-20f);
    for (int it = 0; it < TOPK; ++it) row[idx[it]] += w[it] * scale;
    float* cwr = cw + (size_t)t * NEXP;
    for (int e = 0; e < NEXP; ++e) cwr[e] = row[e];
  }
}

// ---------------------------------------------------------------------------
// Kernel 2: deterministic compaction.  32 waves (one per expert) scan cw
// columns in token order; ballot/popcount gives stable slots.
// ---------------------------------------------------------------------------
__global__ void __launch_bounds__(1024)
k_assign(const float* __restrict__ cw, int* __restrict__ tokids,
         float* __restrict__ wts, int* __restrict__ counts,
         int* __restrict__ offsets) {
  const int tid = threadIdx.x;
  const int e = tid >> 5, lane = tid & 31;
  __shared__ int scnt[NEXP];
  __shared__ int soff[NEXP + 1];
  int cnt = 0;
  for (int t0 = 0; t0 < NTOK; t0 += 32) {
    float v = cw[(size_t)(t0 + lane) * NEXP + e];
    unsigned m = (unsigned)__builtin_amdgcn_ballot_w32(v > 0.f);
    cnt += __popc(m);
  }
  if (lane == 0) scnt[e] = cnt;
  __syncthreads();
  if (tid == 0) {
    int acc = 0;
    for (int i = 0; i < NEXP; ++i) { soff[i] = acc; acc += scnt[i]; }
    soff[NEXP] = acc;
    for (int i = 0; i < NEXP; ++i) { counts[i] = scnt[i]; offsets[i] = soff[i]; }
    offsets[NEXP] = acc;
  }
  __syncthreads();
  int pos = soff[e];
  for (int t0 = 0; t0 < NTOK; t0 += 32) {
    int t = t0 + lane;
    float v = cw[(size_t)t * NEXP + e];
    unsigned m = (unsigned)__builtin_amdgcn_ballot_w32(v > 0.f);
    if (v > 0.f) {
      int sl = pos + __popc(m & ((1u << lane) - 1u));
      tokids[sl] = t;
      wts[sl] = v;
    }
    pos += __popc(m);
  }
}

// ---------------------------------------------------------------------------
// Kernel 3: routed gate_up + SwiGLU -> bf16 intermediate H.
// Block = 256 thr (8 waves) handles 32 gathered tokens x 64 intermediate cols.
// Wave (wrow,wcol): 16x16 gate tile + 16x16 up tile, K swept 2048 in 32-chunks.
// A (tokens) staged in LDS as bf16, B (weights) streamed f32 -> bf16 in regs.
// ---------------------------------------------------------------------------
__global__ void __launch_bounds__(256)
k_gateup_routed(const float* __restrict__ x, const float* __restrict__ gup,
                const int* __restrict__ tokids, const int* __restrict__ counts,
                const int* __restrict__ offsets, bf16_t* __restrict__ H) {
  const int e = blockIdx.y;
  const int cnt = counts[e];
  const int base = blockIdx.x * 32;
  if (base >= cnt) return;
  const int off = offsets[e];
  const int tid = threadIdx.x;
  const int wave = tid >> 5, lane = tid & 31;
  const int wrow = wave & 1, wcol = wave >> 1;
  const int j0 = blockIdx.z * 64 + wcol * 16;
  const int n = lane & 15;
  const int khalf = (lane < 16) ? 0 : 8;

  __shared__ bf16_t As[32 * 72];   // 32 rows, 72-elem stride (16B-aligned, padded)

  const float* gbase = gup + (size_t)e * (2 * MOEI) * HID_;
  const float* gw = gbase + (size_t)(j0 + n) * HID_;
  const float* uw = gbase + (size_t)(MOEI + j0 + n) * HID_;

  // staging role: thread -> (row, 8-wide column chunk)
  const int arow = tid >> 3;
  const int acol = (tid & 7) * 8;
  const float* xr = nullptr;
  if (base + arow < cnt) xr = x + (size_t)tokids[off + base + arow] * HID_;

  v8f accg = {}; v8f accu = {};
  for (int k0 = 0; k0 < HID_; k0 += 64) {
    __syncthreads();
    v8bf st = {};
    if (xr) {
      v8f f = *(const v8f*)(xr + k0 + acol);
      st = __builtin_convertvector(f, v8bf);
    }
    *(v8bf*)&As[arow * 72 + acol] = st;
    __syncthreads();
    const bf16_t* ab = &As[(wrow * 16 + n) * 72];
#pragma unroll
    for (int ks = 0; ks < 2; ++ks) {
      const int kk = ks * 32 + khalf;
      v16bf a  = load_frag_bf16(ab + kk);
      v16bf bg = load_frag_f32(gw + k0 + kk);
      v16bf bu = load_frag_f32(uw + k0 + kk);
      accg = wmma_bf16(a, bg, accg);
      accu = wmma_bf16(a, bu, accu);
    }
  }
  const int mbase = (lane < 16) ? 0 : 8;
#pragma unroll
  for (int r = 0; r < 8; ++r) {
    const int rit = wrow * 16 + mbase + r;
    if (base + rit < cnt) {
      float g = accg[r], u = accu[r];
      float h = silu_f(g) * u;
      H[(size_t)(off + base + rit) * MOEI + j0 + n] = (bf16_t)h;
    }
  }
}

// ---------------------------------------------------------------------------
// Kernel 4: routed down projection + weighted scatter-accumulate into out.
// Block: 32 gathered rows x 64 output cols; K = 768.
// ---------------------------------------------------------------------------
__global__ void __launch_bounds__(256)
k_down_routed(const bf16_t* __restrict__ H, const float* __restrict__ dp,
              const int* __restrict__ tokids, const float* __restrict__ wts,
              const int* __restrict__ counts, const int* __restrict__ offsets,
              float* __restrict__ out) {
  const int e = blockIdx.y;
  const int cnt = counts[e];
  const int base = blockIdx.x * 32;
  if (base >= cnt) return;
  const int off = offsets[e];
  const int tid = threadIdx.x;
  const int wave = tid >> 5, lane = tid & 31;
  const int wrow = wave & 1, wcol = wave >> 1;
  const int n0 = blockIdx.z * 64 + wcol * 16;
  const int n = lane & 15;
  const int khalf = (lane < 16) ? 0 : 8;

  int aslot = off + base + wrow * 16 + n;
  if (aslot > NASSIGN - 1) aslot = NASSIGN - 1;   // clamp tail over-read
  const bf16_t* ar = H + (size_t)aslot * MOEI;
  const float* br = dp + (size_t)e * HID_ * MOEI + (size_t)(n0 + n) * MOEI;

  v8f acc = {};
  for (int k0 = 0; k0 < MOEI; k0 += 32) {
    v16bf a = load_frag_bf16(ar + k0 + khalf);
    v16bf b = load_frag_f32(br + k0 + khalf);
    acc = wmma_bf16(a, b, acc);
  }
  const int mbase = (lane < 16) ? 0 : 8;
#pragma unroll
  for (int r = 0; r < 8; ++r) {
    const int rit = wrow * 16 + mbase + r;
    if (base + rit < cnt) {
      const int slot = off + base + rit;
      const int tok = tokids[slot];
      const float w = wts[slot];
      atomicAdd(out + (size_t)tok * HID_ + n0 + n, w * acc[r]);
    }
  }
}

// ---------------------------------------------------------------------------
// Kernel 5: shared-expert gate_up + SwiGLU (dense over all 2048 tokens).
// ---------------------------------------------------------------------------
__global__ void __launch_bounds__(256)
k_gateup_shared(const float* __restrict__ x, const float* __restrict__ gwp,
                const float* __restrict__ uwp, bf16_t* __restrict__ Hs) {
  const int base = blockIdx.x * 32;
  const int tid = threadIdx.x;
  const int wave = tid >> 5, lane = tid & 31;
  const int wrow = wave & 1, wcol = wave >> 1;
  const int j0 = blockIdx.z * 64 + wcol * 16;
  const int n = lane & 15;
  const int khalf = (lane < 16) ? 0 : 8;

  __shared__ bf16_t As[32 * 72];

  const float* gw = gwp + (size_t)(j0 + n) * HID_;
  const float* uw = uwp + (size_t)(j0 + n) * HID_;
  const int arow = tid >> 3;
  const int acol = (tid & 7) * 8;
  const float* xr = x + (size_t)(base + arow) * HID_;

  v8f accg = {}; v8f accu = {};
  for (int k0 = 0; k0 < HID_; k0 += 64) {
    __syncthreads();
    v8f f = *(const v8f*)(xr + k0 + acol);
    *(v8bf*)&As[arow * 72 + acol] = __builtin_convertvector(f, v8bf);
    __syncthreads();
    const bf16_t* ab = &As[(wrow * 16 + n) * 72];
#pragma unroll
    for (int ks = 0; ks < 2; ++ks) {
      const int kk = ks * 32 + khalf;
      v16bf a = load_frag_bf16(ab + kk);
      accg = wmma_bf16(a, load_frag_f32(gw + k0 + kk), accg);
      accu = wmma_bf16(a, load_frag_f32(uw + k0 + kk), accu);
    }
  }
  const int mbase = (lane < 16) ? 0 : 8;
#pragma unroll
  for (int r = 0; r < 8; ++r) {
    const int rit = wrow * 16 + mbase + r;
    float g = accg[r], u = accu[r];
    Hs[(size_t)(base + rit) * MOEI + j0 + n] = (bf16_t)(silu_f(g) * u);
  }
}

// ---------------------------------------------------------------------------
// Kernel 6: shared-expert down projection.  Plain '=' store: initializes the
// entire output buffer; routed kernel then atomically accumulates on top.
// ---------------------------------------------------------------------------
__global__ void __launch_bounds__(256)
k_down_shared(const bf16_t* __restrict__ Hs, const float* __restrict__ dwp,
              float* __restrict__ out) {
  const int base = blockIdx.x * 32;
  const int tid = threadIdx.x;
  const int wave = tid >> 5, lane = tid & 31;
  const int wrow = wave & 1, wcol = wave >> 1;
  const int n0 = blockIdx.z * 64 + wcol * 16;
  const int n = lane & 15;
  const int khalf = (lane < 16) ? 0 : 8;

  const bf16_t* ar = Hs + (size_t)(base + wrow * 16 + n) * MOEI;
  const float* br = dwp + (size_t)(n0 + n) * MOEI;

  v8f acc = {};
  for (int k0 = 0; k0 < MOEI; k0 += 32) {
    acc = wmma_bf16(load_frag_bf16(ar + k0 + khalf),
                    load_frag_f32(br + k0 + khalf), acc);
  }
  const int mbase = (lane < 16) ? 0 : 8;
#pragma unroll
  for (int r = 0; r < 8; ++r) {
    const int rit = wrow * 16 + mbase + r;
    out[(size_t)(base + rit) * HID_ + n0 + n] = acc[r];
  }
}

// ---------------------------------------------------------------------------
// Launch
// ---------------------------------------------------------------------------
extern "C" void kernel_launch(void* const* d_in, const int* in_sizes, int n_in,
                              void* d_out, int out_size, void* d_ws, size_t ws_size,
                              hipStream_t stream) {
  const float* x    = (const float*)d_in[0];   // [2,1024,2048]
  const float* rw   = (const float*)d_in[1];   // [32,2048]
  const float* bias = (const float*)d_in[2];   // [32]
  const float* gup  = (const float*)d_in[3];   // [32,1536,2048]
  const float* dp   = (const float*)d_in[4];   // [32,2048,768]
  const float* sgw  = (const float*)d_in[5];   // [768,2048]
  const float* suw  = (const float*)d_in[6];   // [768,2048]
  const float* sdw  = (const float*)d_in[7];   // [2048,768]
  float* out = (float*)d_out;

  char* ws = (char*)d_ws;
  float*  cw      = (float*)(ws + WS_CW);
  int*    counts  = (int*)(ws + WS_COUNTS);
  int*    offsets = (int*)(ws + WS_OFFSETS);
  int*    tokids  = (int*)(ws + WS_TOKIDS);
  float*  wts     = (float*)(ws + WS_WTS);
  bf16_t* H       = (bf16_t*)(ws + WS_H);
  bf16_t* Hs      = (bf16_t*)(ws + WS_HS);

  // 1) routing
  k_router<<<NTOK, 32, 0, stream>>>(x, rw, bias, cw);
  // 2) deterministic per-expert compaction
  k_assign<<<1, 1024, 0, stream>>>(cw, tokids, wts, counts, offsets);
  // 3) shared expert (initializes d_out with '=')
  k_gateup_shared<<<dim3(64, 1, 12), 256, 0, stream>>>(x, sgw, suw, Hs);
  k_down_shared<<<dim3(64, 1, 32), 256, 0, stream>>>(Hs, sdw, out);
  // 4) routed experts (accumulate on top with atomics)
  k_gateup_routed<<<dim3(64, NEXP, 12), 256, 0, stream>>>(x, gup, tokids, counts, offsets, H);
  k_down_routed<<<dim3(64, NEXP, 32), 256, 0, stream>>>(H, dp, tokids, wts, counts, offsets, out);
}
